// GroupQueryAttention_65670049956195
// MI455X (gfx1250) — matrix operand order repaired and checked by reading, not compiled, not run on hardware
//
#include <hip/hip_runtime.h>
#include <hip/hip_bf16.h>

#define BDIM 4
#define TSEQ 2048
#define CEMB 2048
#define NQH  16
#define NKV  4
#define HSZ  128
#define MROWS (BDIM * TSEQ)   // 8192

typedef __attribute__((ext_vector_type(16))) __bf16 bf16x16;
typedef __attribute__((ext_vector_type(8)))  __bf16 bf16x8;
typedef __attribute__((ext_vector_type(8)))  float  f32x8;
typedef __attribute__((__vector_size__(16))) int v4i_vs;   // matches builtin param type

#define GLOBAL_AS __attribute__((address_space(1)))
#define LDS_AS    __attribute__((address_space(3)))

union ABfrag { bf16x16 v; bf16x8 h2[2]; };

#ifndef __has_builtin
#define __has_builtin(x) 0
#endif
#if __has_builtin(__builtin_amdgcn_global_load_async_to_lds_b128) && \
    __has_builtin(__builtin_amdgcn_s_wait_asynccnt)
#define USE_ASYNC_LDS 1
#else
#define USE_ASYNC_LDS 0
#endif

static __device__ __forceinline__ __bf16 f2bf(float f) {
  unsigned u = __float_as_uint(f);
  u += 0x7FFFu + ((u >> 16) & 1u);          // round-to-nearest-even
  unsigned short s = (unsigned short)(u >> 16);
  __bf16 b; __builtin_memcpy(&b, &s, 2);
  return b;
}
static __device__ __forceinline__ float bf2f(__bf16 b) {
  unsigned short s; __builtin_memcpy(&s, &b, 2);
  unsigned u = ((unsigned)s) << 16;
  return __uint_as_float(u);
}

// ---------------- f32 -> bf16 convert ----------------
__global__ void f2bf_kernel(const float* __restrict__ in, __bf16* __restrict__ out, int n) {
  int i = blockIdx.x * blockDim.x + threadIdx.x;
  if (i < n) out[i] = f2bf(in[i]);
}

// ---------------- NT GEMM: C[MxN] = A[MxK] * W[NxK]^T ----------------
// block = 256 threads (8 waves); block tile 128(M) x 64(N); wave owns 16 rows x 64 cols.
// The 64x32 W tile is shared by all 8 waves -> stage it in LDS via async loads
// (double-buffered) so B-operand global traffic is fetched once per block.
template <bool OUT_F32>
__global__ __launch_bounds__(256) void gemm_nt_kernel(
    const __bf16* __restrict__ A, const __bf16* __restrict__ W,
    void* __restrict__ Cout, int M, int N, int K) {
  const int lane = threadIdx.x & 31;
  const int wave = threadIdx.x >> 5;
  const int n    = lane & 15;   // lane % 16
  const int hh   = lane >> 4;   // half-wave selector
  const int m0 = blockIdx.y * 128 + wave * 16;
  const int n0 = blockIdx.x * 64;

  const f32x8 zero = {0.f,0.f,0.f,0.f,0.f,0.f,0.f,0.f};
  f32x8 acc[4];
#pragma unroll
  for (int i = 0; i < 4; ++i) acc[i] = zero;

  const __bf16* arow = A + (size_t)(m0 + n) * K;   // A row = m0 + lane%16

#if USE_ASYNC_LDS
  __shared__ __bf16 wtile[2][64 * 32];             // 2 x 4KB double buffer
  const int tid = threadIdx.x;
  const int ldr = tid >> 2;                        // tile row 0..63
  const int ldc = (tid & 3) * 8;                   // 8-element k-chunk (16B)
  // prefetch tile for k = 0 into buffer 0
  __builtin_amdgcn_global_load_async_to_lds_b128(
      (GLOBAL_AS v4i_vs*)(W + (size_t)(n0 + ldr) * K + ldc),
      (LDS_AS v4i_vs*)&wtile[0][ldr * 32 + ldc], 0, 0);

  for (int k = 0; k < K; k += 32) {
    const int buf = (k >> 5) & 1;
    if (k + 32 < K) {
      __builtin_amdgcn_global_load_async_to_lds_b128(
          (GLOBAL_AS v4i_vs*)(W + (size_t)(n0 + ldr) * K + (k + 32) + ldc),
          (LDS_AS v4i_vs*)&wtile[buf ^ 1][ldr * 32 + ldc], 0, 0);
      __builtin_amdgcn_s_wait_asynccnt(1);   // previous tile landed; next stays in flight
    } else {
      __builtin_amdgcn_s_wait_asynccnt(0);
    }
    __syncthreads();

    ABfrag af;
    // A-frag: element e -> K = hh*16 + e
    af.h2[0] = *(const bf16x8*)(arow + k + hh * 16);
    af.h2[1] = *(const bf16x8*)(arow + k + hh * 16 + 8);
#pragma unroll
    for (int nt = 0; nt < 4; ++nt) {
      // B-frag (32x16 KxN): lane holds tile row (nt*16+n), 16 contiguous K at hh*16
      bf16x16 wf = *(const bf16x16*)&wtile[buf][(nt * 16 + n) * 32 + hh * 16];
      acc[nt] = __builtin_amdgcn_wmma_f32_16x16x32_bf16(
          false, af.v, false, wf, (short)0, acc[nt], false, false);
    }
    __syncthreads();   // reads done before this buffer is overwritten at k+64
  }
#else
  for (int k = 0; k < K; k += 32) {
    ABfrag af;
    af.h2[0] = *(const bf16x8*)(arow + k + hh * 16);
    af.h2[1] = *(const bf16x8*)(arow + k + hh * 16 + 8);
#pragma unroll
    for (int nt = 0; nt < 4; ++nt) {
      const __bf16* wrow = W + (size_t)(n0 + nt * 16 + n) * K + k + hh * 16;
      __builtin_prefetch(wrow + 64, 0, 1);
      bf16x16 wf = *(const bf16x16*)wrow;
      acc[nt] = __builtin_amdgcn_wmma_f32_16x16x32_bf16(
          false, af.v, false, wf, (short)0, acc[nt], false, false);
    }
  }
#endif

#pragma unroll
  for (int nt = 0; nt < 4; ++nt) {
#pragma unroll
    for (int j = 0; j < 8; ++j) {
      int row = m0 + j + 8 * hh;          // D layout: lane L vgpr j -> (m=j+8*(L/16), n=L%16)
      int col = n0 + nt * 16 + n;
      float v = acc[nt][j];
      if (OUT_F32) ((float*)Cout)[(size_t)row * N + col] = v;
      else         ((__bf16*)Cout)[(size_t)row * N + col] = f2bf(v);
    }
  }
}

// ---------------- RoPE (in place on bf16 Q or K buffer) ----------------
__global__ void rope_kernel(__bf16* __restrict__ X, int nheads) {
  int gid = blockIdx.x;            // (b*T + t)*nheads + h
  int h   = gid % nheads;
  int bt  = gid / nheads;
  int t   = bt % TSEQ;
  int d   = threadIdx.x;           // 0..63
  __bf16* base = X + (size_t)bt * (nheads * HSZ) + h * HSZ;
  float x1 = bf2f(base[d]);
  float x2 = bf2f(base[d + 64]);
  float freq = __powf(10000.0f, -(float)d * (1.0f / 64.0f));
  float s, c;
  __sincosf((float)t * freq, &s, &c);
  base[d]      = f2bf(x1 * c - x2 * s);
  base[d + 64] = f2bf(x2 * c + x1 * s);
}

// ---------------- V transpose: (b,t,kvh,d) -> (b,kvh,d,t) ----------------
__global__ void vtrans_kernel(const __bf16* __restrict__ V, __bf16* __restrict__ Vt) {
  int i = blockIdx.x * blockDim.x + threadIdx.x;
  if (i >= MROWS * NKV * HSZ) return;
  int d   = i & (NKV * HSZ - 1);
  int row = i >> 9;                 // b*T + t
  int b = row / TSEQ, t = row % TSEQ;
  int kvh = d >> 7, dd = d & 127;
  Vt[(((size_t)b * NKV + kvh) * HSZ + dd) * TSEQ + t] = V[i];
}

// ---------------- Flash attention (causal, online softmax) ----------------
// grid = (T/128, B*NQH); 8 waves/block; each wave: one 16-row query tile, key steps of 32.
__global__ __launch_bounds__(256) void flash_kernel(
    const __bf16* __restrict__ Q, const __bf16* __restrict__ Kc,
    const __bf16* __restrict__ Vt, __bf16* __restrict__ Oa) {
  __shared__ __bf16 plds[8 * 512];            // 1KB (16x32 bf16) per wave
  const int lane = threadIdx.x & 31;
  const int wave = threadIdx.x >> 5;
  const int n  = lane & 15;
  const int hh = lane >> 4;
  const int qt = blockIdx.x * 8 + wave;       // query tile index (16 rows)
  const int bh = blockIdx.y;
  const int b  = bh / NQH;
  const int h  = bh % NQH;
  const int kvh = h / (NQH / NKV);
  __bf16* wlds = plds + wave * 512;

  // Preload Q fragments: 16x128 as 4 chunks of K=32 (A-operand layout)
  ABfrag qf[4];
  const __bf16* qrow = Q + ((size_t)(b * TSEQ + qt * 16 + n)) * (NQH * HSZ) + h * HSZ;
#pragma unroll
  for (int kk = 0; kk < 4; ++kk) {
    qf[kk].h2[0] = *(const bf16x8*)(qrow + kk * 32 + hh * 16);
    qf[kk].h2[1] = *(const bf16x8*)(qrow + kk * 32 + hh * 16 + 8);
  }

  const f32x8 zero = {0.f,0.f,0.f,0.f,0.f,0.f,0.f,0.f};
  f32x8 O[8];
#pragma unroll
  for (int i = 0; i < 8; ++i) O[i] = zero;
  float m_i[8], l_i[8];
#pragma unroll
  for (int j = 0; j < 8; ++j) { m_i[j] = -3.0e38f; l_i[j] = 0.0f; }

  const float scale = 0.08838834764831845f;   // 1/sqrt(128)
  const int qmax = qt * 16 + 15;
  const __bf16* kbase = Kc + (size_t)(b * TSEQ) * (NKV * HSZ) + kvh * HSZ;
  const __bf16* vbase = Vt + ((size_t)(b * NKV + kvh) * HSZ) * TSEQ;

  for (int j0 = 0; j0 <= qmax; j0 += 32) {
    // S = Q K^T over head dim (K=128 -> 4 WMMA per 16x16 S tile)
    f32x8 S0 = zero, S1 = zero;
#pragma unroll
    for (int kk = 0; kk < 4; ++kk) {
      bf16x16 b0 = *(const bf16x16*)(kbase + (size_t)(j0 + n) * (NKV * HSZ) + kk * 32 + hh * 16);
      S0 = __builtin_amdgcn_wmma_f32_16x16x32_bf16(false, qf[kk].v, false, b0, (short)0, S0, false, false);
    }
#pragma unroll
    for (int kk = 0; kk < 4; ++kk) {
      bf16x16 b1 = *(const bf16x16*)(kbase + (size_t)(j0 + 16 + n) * (NKV * HSZ) + kk * 32 + hh * 16);
      S1 = __builtin_amdgcn_wmma_f32_16x16x32_bf16(false, qf[kk].v, false, b1, (short)0, S1, false, false);
    }
    // scale + causal mask (D layout: row m = j + 8*hh, key = j0 [+16] + n)
#pragma unroll
    for (int j = 0; j < 8; ++j) {
      int q = qt * 16 + j + 8 * hh;
      float s0 = S0[j] * scale;
      float s1 = S1[j] * scale;
      S0[j] = (j0 + n      <= q) ? s0 : -3.0e38f;
      S1[j] = (j0 + 16 + n <= q) ? s1 : -3.0e38f;
    }
    // online softmax: row reductions across the 16 lanes of each half-wave
    float alpha[8];
#pragma unroll
    for (int j = 0; j < 8; ++j) {
      float tm = fmaxf(S0[j], S1[j]);
      tm = fmaxf(tm, __shfl_xor(tm, 1, 32));
      tm = fmaxf(tm, __shfl_xor(tm, 2, 32));
      tm = fmaxf(tm, __shfl_xor(tm, 4, 32));
      tm = fmaxf(tm, __shfl_xor(tm, 8, 32));
      float mn = fmaxf(m_i[j], tm);
      alpha[j] = __expf(m_i[j] - mn);
      m_i[j] = mn;
      float p0 = __expf(S0[j] - mn);
      float p1 = __expf(S1[j] - mn);
      S0[j] = p0; S1[j] = p1;
      float rs = p0 + p1;
      rs += __shfl_xor(rs, 1, 32);
      rs += __shfl_xor(rs, 2, 32);
      rs += __shfl_xor(rs, 4, 32);
      rs += __shfl_xor(rs, 8, 32);
      l_i[j] = l_i[j] * alpha[j] + rs;
    }
#pragma unroll
    for (int nt = 0; nt < 8; ++nt)
#pragma unroll
      for (int j = 0; j < 8; ++j)
        O[nt][j] *= alpha[j];

    // P: D layout -> A layout via per-wave LDS staging (LDS is in-order per wave)
#pragma unroll
    for (int j = 0; j < 8; ++j) {
      wlds[(j + 8 * hh) * 32 + n]      = f2bf(S0[j]);
      wlds[(j + 8 * hh) * 32 + 16 + n] = f2bf(S1[j]);
    }
    ABfrag pa;
    pa.h2[0] = *(const bf16x8*)(wlds + n * 32 + hh * 16);
    pa.h2[1] = *(const bf16x8*)(wlds + n * 32 + hh * 16 + 8);

    // O += P * V  (contraction over 32 keys; V pre-transposed so B loads are contiguous)
#pragma unroll
    for (int nt = 0; nt < 8; ++nt) {
      bf16x16 vb = *(const bf16x16*)(vbase + (size_t)(nt * 16 + n) * TSEQ + j0 + hh * 16);
      O[nt] = __builtin_amdgcn_wmma_f32_16x16x32_bf16(false, pa.v, false, vb, (short)0, O[nt], false, false);
    }
  }
  // normalize and store bf16 into (B*T, 2048) for the out-projection GEMM
#pragma unroll
  for (int j = 0; j < 8; ++j) {
    float inv = 1.0f / l_i[j];
    int q = qt * 16 + j + 8 * hh;
    __bf16* orow = Oa + ((size_t)(b * TSEQ + q)) * (NQH * HSZ) + h * HSZ;
#pragma unroll
    for (int nt = 0; nt < 8; ++nt)
      orow[nt * 16 + n] = f2bf(O[nt][j] * inv);
  }
}

// ---------------- host launcher ----------------
extern "C" void kernel_launch(void* const* d_in, const int* in_sizes, int n_in,
                              void* d_out, int out_size, void* d_ws, size_t ws_size,
                              hipStream_t stream) {
  const float* x  = (const float*)d_in[0];
  const float* Wq = (const float*)d_in[1];
  const float* Wk = (const float*)d_in[2];
  const float* Wv = (const float*)d_in[3];
  const float* Wp = (const float*)d_in[4];

  char* ws = (char*)d_ws;
  size_t off = 0;
  #define WSALLOC(nbytes) (ws + off); off += (((size_t)(nbytes)) + 255) & ~(size_t)255
  __bf16* xb   = (__bf16*)WSALLOC((size_t)MROWS * CEMB * 2);
  __bf16* Wq16 = (__bf16*)WSALLOC((size_t)CEMB * CEMB * 2);
  __bf16* Wk16 = (__bf16*)WSALLOC((size_t)(NKV * HSZ) * CEMB * 2);
  __bf16* Wv16 = (__bf16*)WSALLOC((size_t)(NKV * HSZ) * CEMB * 2);
  __bf16* Wp16 = (__bf16*)WSALLOC((size_t)CEMB * CEMB * 2);
  __bf16* Qb   = (__bf16*)WSALLOC((size_t)MROWS * (NQH * HSZ) * 2);
  __bf16* Kb   = (__bf16*)WSALLOC((size_t)MROWS * (NKV * HSZ) * 2);
  __bf16* Vb   = (__bf16*)WSALLOC((size_t)MROWS * (NKV * HSZ) * 2);
  __bf16* Vt   = (__bf16*)WSALLOC((size_t)MROWS * (NKV * HSZ) * 2);
  __bf16* At   = (__bf16*)WSALLOC((size_t)MROWS * (NQH * HSZ) * 2);
  #undef WSALLOC

  // 1) f32 -> bf16 conversions
  int n;
  n = MROWS * CEMB;            f2bf_kernel<<<(n + 255) / 256, 256, 0, stream>>>(x,  xb,   n);
  n = CEMB * CEMB;             f2bf_kernel<<<(n + 255) / 256, 256, 0, stream>>>(Wq, Wq16, n);
  n = NKV * HSZ * CEMB;        f2bf_kernel<<<(n + 255) / 256, 256, 0, stream>>>(Wk, Wk16, n);
  n = NKV * HSZ * CEMB;        f2bf_kernel<<<(n + 255) / 256, 256, 0, stream>>>(Wv, Wv16, n);
  n = CEMB * CEMB;             f2bf_kernel<<<(n + 255) / 256, 256, 0, stream>>>(Wp, Wp16, n);

  // 2) QKV projections (WMMA NT-GEMM, bf16 out)
  gemm_nt_kernel<false><<<dim3(CEMB / 64, MROWS / 128), 256, 0, stream>>>(
      xb, Wq16, Qb, MROWS, CEMB, CEMB);
  gemm_nt_kernel<false><<<dim3((NKV * HSZ) / 64, MROWS / 128), 256, 0, stream>>>(
      xb, Wk16, Kb, MROWS, NKV * HSZ, CEMB);
  gemm_nt_kernel<false><<<dim3((NKV * HSZ) / 64, MROWS / 128), 256, 0, stream>>>(
      xb, Wv16, Vb, MROWS, NKV * HSZ, CEMB);

  // 3) RoPE on Q and K; V transpose for contiguous PV loads
  rope_kernel<<<BDIM * TSEQ * NQH, 64, 0, stream>>>(Qb, NQH);
  rope_kernel<<<BDIM * TSEQ * NKV, 64, 0, stream>>>(Kb, NKV);
  n = MROWS * NKV * HSZ;
  vtrans_kernel<<<(n + 255) / 256, 256, 0, stream>>>(Vb, Vt);

  // 4) causal flash attention
  flash_kernel<<<dim3(TSEQ / 128, BDIM * NQH), 256, 0, stream>>>(Qb, Kb, Vt, At);

  // 5) output projection (f32 out straight into d_out)
  gemm_nt_kernel<true><<<dim3(CEMB / 64, MROWS / 128), 256, 0, stream>>>(
      At, Wp16, (float*)d_out, MROWS, CEMB, CEMB);
}
